// AttentionNet_39591008534866
// MI455X (gfx1250) — matrix operand order repaired and checked by reading, not compiled, not run on hardware
//
#include <hip/hip_runtime.h>
#include <cstddef>

// ---------------- problem constants ----------------
#define Bsz 256
#define Nn  400
#define Kk  20
#define Dm  128
#define FFd 512
#define NH  4
#define DHd 32
#define LKP 416                     // keys padded to multiple of 32
#define MROWS (Bsz*Nn)              // 102400

typedef __attribute__((ext_vector_type(16))) _Float16 v16h;
typedef __attribute__((ext_vector_type(8)))  float    v8f;

#ifndef __has_builtin
#define __has_builtin(x) 0
#endif
#if __has_builtin(__builtin_amdgcn_global_load_async_to_lds_b128) && \
    __has_builtin(__builtin_amdgcn_s_wait_asynccnt)
#define HAVE_ASYNC_LDS 1
#else
#define HAVE_ASYNC_LDS 0
#endif

// builtin expects: (int __vector(4) AS1*, int __vector(4) AS3*, imm int, imm int)
typedef int v4i_ __attribute__((vector_size(16)));
typedef __attribute__((address_space(1))) v4i_ g_v4i;
typedef __attribute__((address_space(3))) v4i_ l_v4i;

// 16B global -> LDS async copy (per-lane); falls back to a synchronous copy
__device__ __forceinline__ void async_ld16(const void* g, void* l){
#if HAVE_ASYNC_LDS
  __builtin_amdgcn_global_load_async_to_lds_b128(
      (g_v4i*)(unsigned long long)g,
      (l_v4i*)(unsigned int)(unsigned long long)l, 0, 0);
#else
  *(float4*)l = *(const float4*)g;
#endif
}

__device__ __forceinline__ void async_wait0(){
#if HAVE_ASYNC_LDS
  __builtin_amdgcn_s_wait_asynccnt(0);
#endif
}

__device__ __forceinline__ v8f zero_v8(){
  v8f z;
#pragma unroll
  for (int i=0;i<8;++i) z[i]=0.f;
  return z;
}

__device__ __forceinline__ v8f wmma16(v16h a, v16h b, v8f c){
  // D = A(16x32 f16) * B(32x16 f16) + C(16x16 f32)
  return __builtin_amdgcn_wmma_f32_16x16x32_f16(false, a, false, b, (short)0, c, false, false);
}

__device__ __forceinline__ float sigmoidf_(float x){ return 1.f/(1.f+__expf(-x)); }

// ---------------- tiny utility kernels ----------------
__global__ void k_cvt(const float* __restrict__ s, _Float16* __restrict__ d, int n){
  int i = blockIdx.x*blockDim.x + threadIdx.x;
  if (i < n) d[i] = (_Float16)s[i];
}

__global__ void k_zero_h(_Float16* __restrict__ p, size_t n){
  size_t i = (size_t)blockIdx.x*blockDim.x + threadIdx.x;
  if (i < n) p[i] = (_Float16)0.f;
}

// ---------------- embedding: emb = (node@We/Wi + be/bi) + pos@Wp + bp ----------------
__global__ void k_embed(const float* __restrict__ node, const float* __restrict__ pos,
                        const float* __restrict__ We, const float* __restrict__ be,
                        const float* __restrict__ Wi, const float* __restrict__ bi,
                        const float* __restrict__ Wp, const float* __restrict__ bp,
                        float* __restrict__ out){
  size_t idx = (size_t)blockIdx.x*blockDim.x + threadIdx.x;
  if (idx >= (size_t)MROWS*Dm) return;
  int d = (int)(idx & (Dm-1));
  size_t bn = idx >> 7;
  int n = (int)(bn % Nn);
  const float* x  = node + bn*4;
  const float* W  = (n==0)? We : Wi;
  const float* bb = (n==0)? be : bi;
  float acc = bb[d] + bp[d];
#pragma unroll
  for (int k=0;k<4;++k)  acc += x[k]*W[k*Dm+d];
  const float* pe = pos + bn*32;
#pragma unroll
  for (int k=0;k<32;++k) acc += pe[k]*Wp[k*Dm+d];
  out[idx] = acc;
}

// ---------------- generic WMMA GEMM: C = act(A[M,K] @ W[K,N] + bias) ----------------
// Wave = 16x64 tile (4 accumulators share one A fragment).
// Block = 8 waves; the block's A rows are staged in LDS via async global->LDS copies,
// double-buffered over the K loop (ASYNCcnt + barrier).
template<typename AT>
__global__ __launch_bounds__(256) void k_gemm(const AT* __restrict__ A,
                                              const _Float16* __restrict__ W,
                                              const float* __restrict__ bias,
                                              float* __restrict__ Cf32,
                                              _Float16* __restrict__ Cf16,
                                              int M, int Kd, int Nc, int relu){
  extern __shared__ char smem[];           // 2 * rowsPB * 32 * sizeof(AT)
  const int tid  = threadIdx.x;
  const int lane = tid & 31;
  const int wv   = tid >> 5;
  const int nt64 = Nc >> 6;                // 64-col tiles per m-row (2 or 8)
  const int id   = blockIdx.x*8 + wv;
  const int mt   = id / nt64;
  const int nt   = id % nt64;
  const int mtBase = (blockIdx.x*8) / nt64;
  const int m0   = mt << 4;
  const int n0   = nt << 6;
  const int rowsPB = 128 / nt64;           // rows staged per block (64 or 16)
  const size_t mrow0 = (size_t)mtBase << 4;
  AT* abuf = (AT*)smem;
  const int chunkElems = rowsPB * 32;      // one K-step of staged A
  const int rowBytes   = 32 * (int)sizeof(AT);
  const int nchunk     = (chunkElems * (int)sizeof(AT)) >> 4;

  // prologue: stage K-chunk 0 into buffer 0
  for (int c = tid; c < nchunk; c += 256){
    int row = (c << 4) / rowBytes;
    int inb = (c << 4) % rowBytes;
    const char* g = (const char*)(A + (mrow0 + row) * Kd) + inb;
    async_ld16(g, (char*)abuf + (size_t)c * 16);
  }

  const int grp  = lane >> 4;
  const int koff = grp ? 8 : 0;
  const int arow = ((mt - mtBase) << 4) + (lane & 15);
  v8f acc[4];
#pragma unroll
  for (int j=0;j<4;++j) acc[j] = zero_v8();
  const int iters = Kd >> 5;
  for (int it = 0; it < iters; ++it){
    const int buf = it & 1;
    async_wait0();                          // my staged chunk has landed
    __syncthreads();                        // everyone's chunk landed; prev reads done
    if (it + 1 < iters){                    // prefetch next K-chunk into other buffer
      AT* nb2 = abuf + (size_t)(buf ^ 1) * chunkElems;
      const int kb2 = (it + 1) << 5;
      for (int c = tid; c < nchunk; c += 256){
        int row = (c << 4) / rowBytes;
        int inb = (c << 4) % rowBytes;
        const char* g = (const char*)(A + (mrow0 + row) * Kd + kb2) + inb;
        async_ld16(g, (char*)nb2 + (size_t)c * 16);
      }
    }
    const AT* ar = abuf + (size_t)buf * chunkElems + (size_t)arow * 32;
    v16h a;
#pragma unroll
    for (int v=0; v<8; ++v){
      int k = koff + ((v&3)<<1) + ((v&4)?16:0);
      a[2*v]   = (_Float16)ar[k];
      a[2*v+1] = (_Float16)ar[k+1];
    }
    const int kb = it << 5;
    const _Float16* wrow = W + (size_t)(kb + lane)*Nc + n0;
    if (kb + 32 < Kd) __builtin_prefetch(wrow + 32*Nc, 0, 0);   // global_prefetch_b8
    v16h bm[4];
#pragma unroll
    for (int j=0;j<4;++j){
#pragma unroll
      for (int v=0; v<8; ++v){
        int k = kb + koff + ((v&3)<<1) + ((v&4)?16:0);
        bm[j][2*v]   = W[(size_t)k*Nc + n0 + (j<<4) + (lane&15)];
        bm[j][2*v+1] = W[(size_t)(k+1)*Nc + n0 + (j<<4) + (lane&15)];
      }
    }
#pragma unroll
    for (int j=0;j<4;++j) acc[j] = wmma16(a, bm[j], acc[j]);
  }
#pragma unroll
  for (int j=0;j<4;++j){
    const int n = n0 + (j<<4) + (lane & 15);
    float bv = bias ? bias[n] : 0.f;
#pragma unroll
    for (int r=0;r<8;++r){
      int m = m0 + r + (grp<<3);
      float val = acc[j][r] + bv;
      if (relu) val = fmaxf(val, 0.f);
      size_t o = (size_t)m*Nc + n;
      if (Cf32) Cf32[o] = val;
      if (Cf16) Cf16[o] = (_Float16)val;
    }
  }
}

// ---------------- QKV projection with head-split scatter (f16 out) ----------------
// Wave = 16x64 tile (4 accumulators, one A fragment).
// mode 0: Q[b,h,q,d]   mode 1: Kt[b,h,d,k]  (transposed, padded)   mode 2: V[b,h,k,d] (padded)
__global__ __launch_bounds__(256) void k_gemm_qkv(const float* __restrict__ A,
                                                  const _Float16* __restrict__ W,
                                                  _Float16* __restrict__ Out, int mode){
  const int lane = threadIdx.x & 31;
  const int id   = blockIdx.x*8 + (threadIdx.x >> 5);
  const int m0   = (id >> 1) << 4;         // Nc = 128 -> 2 n64-tiles
  const int n0   = (id &  1) << 6;
  const int grp  = lane >> 4;
  const int koff = grp ? 8 : 0;
  const float* arow = A + (size_t)(m0 + (lane&15))*Dm;
  v8f acc[4];
#pragma unroll
  for (int j=0;j<4;++j) acc[j] = zero_v8();
#pragma unroll
  for (int kb=0; kb<Dm; kb+=32){
    v16h a;
#pragma unroll
    for (int v=0; v<8; ++v){
      int k = kb + koff + ((v&3)<<1) + ((v&4)?16:0);
      a[2*v]   = (_Float16)arow[k];
      a[2*v+1] = (_Float16)arow[k+1];
    }
    const _Float16* wrow = W + (size_t)(kb + lane)*Dm + n0;
#pragma unroll
    for (int j=0;j<4;++j){
      v16h bw;
#pragma unroll
      for (int v=0; v<8; ++v){
        int k = kb + koff + ((v&3)<<1) + ((v&4)?16:0);
        bw[2*v]   = W[(size_t)k*Dm + n0 + (j<<4) + (lane&15)];
        bw[2*v+1] = W[(size_t)(k+1)*Dm + n0 + (j<<4) + (lane&15)];
      }
      acc[j] = wmma16(a, bw, acc[j]);
    }
  }
#pragma unroll
  for (int j=0;j<4;++j){
    int n = n0 + (j<<4) + (lane & 15);
    int h = n >> 5, d = n & 31;
#pragma unroll
    for (int r=0;r<8;++r){
      int m = m0 + r + (grp<<3);
      int b = m / Nn, nq = m % Nn;
      size_t bh = (size_t)b*NH + h;
      size_t o;
      if (mode == 0)      o = (bh*Nn  + nq)*DHd + d;
      else if (mode == 1) o = (bh*DHd + d )*LKP + nq;
      else                o = (bh*LKP + nq)*DHd + d;
      Out[o] = (_Float16)acc[j][r];
    }
  }
}

// ---------------- flash attention: one 16-query tile per wave ----------------
__global__ __launch_bounds__(256) void k_attn(const _Float16* __restrict__ Qh,
                                              const _Float16* __restrict__ Kt,
                                              const _Float16* __restrict__ Vp,
                                              float* __restrict__ O){
  __shared__ _Float16 lp[8][16*32];       // per-wave P staging tile
  int lane = threadIdx.x & 31;
  int wv   = threadIdx.x >> 5;
  int w    = blockIdx.x*8 + wv;           // [0, B*H*25)
  int qt   = w % 25;
  int bh   = w / 25;
  int b = bh >> 2, h = bh & 3;
  const _Float16* Qb = Qh + ((size_t)bh*Nn + qt*16)*DHd;
  const _Float16* Kb = Kt + (size_t)bh*DHd*LKP;
  const _Float16* Vb = Vp + (size_t)bh*LKP*DHd;
  _Float16* myp = lp[wv];
  int m    = lane & 15;
  int grp  = lane >> 4;
  int koff = grp ? 8 : 0;
  // A-operand: Q tile (dh = 32 = full contraction), loaded once
  const _Float16* qrow = Qb + m*DHd;
  v16h aq;
#pragma unroll
  for (int v=0; v<8; ++v){
    int k = koff + ((v&3)<<1) + ((v&4)?16:0);
    aq[2*v] = qrow[k]; aq[2*v+1] = qrow[k+1];
  }
  float mrun[8], lrun[8];
#pragma unroll
  for (int r=0;r<8;++r){ mrun[r] = -3e38f; lrun[r] = 0.f; }
  v8f o0 = zero_v8(), o1 = zero_v8();
  const float sc = 0.17677669529663687f;  // 1/sqrt(32)
  for (int jb=0; jb<LKP; jb+=32){
    v16h bk0, bk1;
#pragma unroll
    for (int v=0; v<8; ++v){
      int k = koff + ((v&3)<<1) + ((v&4)?16:0);
      bk0[2*v]   = Kb[(size_t)k*LKP + jb + (lane&15)];
      bk0[2*v+1] = Kb[(size_t)(k+1)*LKP + jb + (lane&15)];
      bk1[2*v]   = Kb[(size_t)k*LKP + jb + 16 + (lane&15)];
      bk1[2*v+1] = Kb[(size_t)(k+1)*LKP + jb + 16 + (lane&15)];
    }
    v8f z = zero_v8();
    v8f s0 = wmma16(aq, bk0, z);
    v8f s1 = wmma16(aq, bk1, z);
    int key0 = jb + (lane & 15), key1 = key0 + 16;
    float p0[8], p1[8], alpha[8];
#pragma unroll
    for (int r=0;r<8;++r){
      float a0 = (key0 < Nn) ? s0[r]*sc : -3e38f;
      float a1 = (key1 < Nn) ? s1[r]*sc : -3e38f;
      float mx = fmaxf(a0, a1);
#pragma unroll
      for (int off=1; off<16; off<<=1) mx = fmaxf(mx, __shfl_xor(mx, off));
      float mnew = fmaxf(mrun[r], mx);
      float al = __expf(mrun[r] - mnew);
      float e0 = __expf(a0 - mnew);
      float e1 = __expf(a1 - mnew);
      float ls = e0 + e1;
#pragma unroll
      for (int off=1; off<16; off<<=1) ls += __shfl_xor(ls, off);
      lrun[r] = lrun[r]*al + ls;
      mrun[r] = mnew;
      alpha[r] = al; p0[r] = e0; p1[r] = e1;
    }
#pragma unroll
    for (int r=0;r<8;++r){ o0[r] *= alpha[r]; o1[r] *= alpha[r]; }
    // C-layout -> LDS (f16) -> A-layout; wave-private tile, LDS is in-order per wave
#pragma unroll
    for (int r=0;r<8;++r){
      int row = r + (grp<<3);
      myp[row*32 + (lane&15)]      = (_Float16)p0[r];
      myp[row*32 + (lane&15) + 16] = (_Float16)p1[r];
    }
    __builtin_amdgcn_wave_barrier();
    v16h ap;
    const _Float16* prow = myp + m*32;
#pragma unroll
    for (int v=0; v<8; ++v){
      int k = koff + ((v&3)<<1) + ((v&4)?16:0);
      ap[2*v] = prow[k]; ap[2*v+1] = prow[k+1];
    }
    __builtin_amdgcn_wave_barrier();
    v16h bv0, bv1;
#pragma unroll
    for (int v=0; v<8; ++v){
      int k = koff + ((v&3)<<1) + ((v&4)?16:0);
      bv0[2*v]   = Vb[(size_t)(jb + k)*DHd + (lane&15)];
      bv0[2*v+1] = Vb[(size_t)(jb + k + 1)*DHd + (lane&15)];
      bv1[2*v]   = Vb[(size_t)(jb + k)*DHd + 16 + (lane&15)];
      bv1[2*v+1] = Vb[(size_t)(jb + k + 1)*DHd + 16 + (lane&15)];
    }
    o0 = wmma16(ap, bv0, o0);
    o1 = wmma16(ap, bv1, o1);
  }
#pragma unroll
  for (int r=0;r<8;++r){
    float inv = 1.f / lrun[r];
    int q = qt*16 + r + (grp<<3);
    float* orow = O + ((size_t)b*Nn + q)*Dm + h*DHd;
    orow[lane & 15]        = o0[r]*inv;
    orow[(lane & 15) + 16] = o1[r]*inv;
  }
}

// ---------------- residual add + LayerNorm, one wave per row ----------------
__global__ __launch_bounds__(256) void k_add_ln(const float* __restrict__ X, const float* __restrict__ Y,
                                                const float* __restrict__ g, const float* __restrict__ bta,
                                                float* __restrict__ Out){
  int lane = threadIdx.x & 31;
  size_t row = (size_t)blockIdx.x*8 + (threadIdx.x >> 5);
  size_t base = row*Dm + lane*4;
  float x[4]; float s=0.f, s2=0.f;
#pragma unroll
  for (int i=0;i<4;++i){ x[i] = X[base+i] + Y[base+i]; s += x[i]; s2 += x[i]*x[i]; }
#pragma unroll
  for (int off=1; off<32; off<<=1){ s += __shfl_xor(s, off); s2 += __shfl_xor(s2, off); }
  float mean = s * (1.f/Dm);
  float var  = s2 * (1.f/Dm) - mean*mean;
  float rstd = rsqrtf(var + 1e-5f);
#pragma unroll
  for (int i=0;i<4;++i){ int c = lane*4+i; Out[base+i] = g[c]*(x[i]-mean)*rstd + bta[c]; }
}

// ---------------- Wb tail: += budget*Wb[128,:] + 0.4*Wb[129,:] ----------------
__global__ void k_wb_tail(float* __restrict__ emb2, const float* __restrict__ budget,
                          const float* __restrict__ Wb){
  size_t idx = (size_t)blockIdx.x*blockDim.x + threadIdx.x;
  if (idx >= (size_t)MROWS*Dm) return;
  int d = (int)(idx & (Dm-1));
  size_t bn = idx >> 7;
  emb2[idx] += budget[bn]*Wb[128*Dm + d] + 0.4f*Wb[129*Dm + d];
}

// ---------------- LSTM cell, one block per batch ----------------
__global__ __launch_bounds__(512) void k_lstm(const float* __restrict__ emb2, const int* __restrict__ cidx,
                                              const float* __restrict__ nb, const float* __restrict__ h0,
                                              const float* __restrict__ c0,
                                              const float* __restrict__ Wih, const float* __restrict__ Whh,
                                              const float* __restrict__ bih, const float* __restrict__ bhh,
                                              float* __restrict__ h1, float* __restrict__ c1){
  int b = blockIdx.x, t = threadIdx.x;
  __shared__ float xv[256], hh[Dm], cc[Dm], gates[512];
  int ci = cidx[b];
  if (t < Dm){ xv[t] = emb2[((size_t)b*Nn + ci)*Dm + t]; hh[t] = h0[b*Dm+t]; cc[t] = c0[b*Dm+t]; }
  else if (t < 256){ xv[t] = nb[b*Dm + (t-128)]; }
  __syncthreads();
  float g = bih[t] + bhh[t];
  const float* wi = Wih + (size_t)t*256;
  for (int k=0;k<256;++k) g += xv[k]*wi[k];
  const float* wh = Whh + (size_t)t*Dm;
  for (int k=0;k<Dm;++k) g += hh[k]*wh[k];
  gates[t] = g;
  __syncthreads();
  if (t < Dm){
    float ig = gates[t], fg = gates[Dm+t], gg = gates[2*Dm+t], og = gates[3*Dm+t];
    float cv = sigmoidf_(fg)*cc[t] + sigmoidf_(ig)*tanhf(gg);
    float hv = sigmoidf_(og)*tanhf(cv);
    c1[b*Dm+t] = cv; h1[b*Dm+t] = hv;
  }
}

// ---------------- decoder block + pointer + value, one block per batch ----------------
__global__ __launch_bounds__(512) void k_decoder(
    const float* __restrict__ emb2, const float* __restrict__ h1,
    const int* __restrict__ edge, const int* __restrict__ cidx,
    const float* __restrict__ budget, const int* __restrict__ maskp,
    const float* __restrict__ Wc, const float* __restrict__ bc,
    const float* __restrict__ Wq, const float* __restrict__ Wk,
    const float* __restrict__ Wv, const float* __restrict__ Wo,
    const float* __restrict__ g1, const float* __restrict__ b1,
    const float* __restrict__ W1, const float* __restrict__ bf1,
    const float* __restrict__ W2, const float* __restrict__ bf2,
    const float* __restrict__ g2, const float* __restrict__ b2,
    const float* __restrict__ Pq, const float* __restrict__ Pk,
    const float* __restrict__ Wval, const float* __restrict__ bval,
    float* __restrict__ logp, float* __restrict__ value)
{
  int b = blockIdx.x, t = threadIdx.x;
  __shared__ float cf[Kk][Dm];
  __shared__ float Khs[Kk][Dm];
  __shared__ float Vhs[Kk][Dm];
  __shared__ float pks[Kk][Dm];
  __shared__ float curs[Dm], qv[Dm], att[Dm], hb[Dm], cfp[Dm], pqs[Dm];
  __shared__ float ffs[FFd];
  __shared__ float prob[NH][Kk];
  __shared__ float cmv[Kk], uarr[Kk];
  __shared__ float red[2];
  int ci = cidx[b];
  for (int i=t; i<Kk*Dm; i+=512){
    int k = i >> 7, d = i & 127;
    int ce = edge[((size_t)b*Nn + ci)*Kk + k];
    cf[k][d] = emb2[((size_t)b*Nn + ce)*Dm + d];
  }
  if (t < Kk){
    int ce = edge[((size_t)b*Nn + ci)*Kk + t];
    float bud = budget[(size_t)b*Nn + ce];
    float c = (bud > 0.f) ? (float)maskp[((size_t)b*Nn + ci)*Kk + t] : 1.f;
    if (t == 0) c = 1.f;
    cmv[t] = c;
  }
  __syncthreads();
  if (t < Dm){                                // cur = [h1, emb2[b,0]] @ Wc + bc
    float a = bc[t];
    for (int k=0;k<Dm;++k) a += h1[b*Dm+k]*Wc[k*Dm+t];
    const float* e0 = emb2 + (size_t)b*Nn*Dm;
    for (int k=0;k<Dm;++k) a += e0[k]*Wc[(Dm+k)*Dm+t];
    curs[t] = a;
  }
  __syncthreads();
  if (t < Dm){
    float a = 0.f;
    for (int j=0;j<Dm;++j) a += curs[j]*Wq[j*Dm+t];
    qv[t] = a;
  }
  for (int i=t; i<Kk*Dm; i+=512){
    int k = i >> 7, d = i & 127;
    float a=0.f, v=0.f;
    for (int j=0;j<Dm;++j){ float c = cf[k][j]; a += c*Wk[j*Dm+d]; v += c*Wv[j*Dm+d]; }
    Khs[k][d]=a; Vhs[k][d]=v;
  }
  __syncthreads();
  if (t < NH*Kk){
    int h = t / Kk, k = t % Kk;
    float s = 0.f;
    for (int d=0;d<DHd;++d) s += qv[h*DHd+d]*Khs[k][h*DHd+d];
    s *= 0.17677669529663687f;
    if (cmv[k] == 1.f) s = -1e8f;
    prob[h][k] = s;
  }
  __syncthreads();
  if (t < NH){
    float mx = -3e38f;
    for (int k=0;k<Kk;++k) mx = fmaxf(mx, prob[t][k]);
    float sum = 0.f;
    for (int k=0;k<Kk;++k){ float e = __expf(prob[t][k]-mx); prob[t][k]=e; sum+=e; }
    float inv = 1.f/sum;
    for (int k=0;k<Kk;++k) prob[t][k] *= inv;
  }
  __syncthreads();
  if (t < Dm){
    int h = t >> 5;
    float a = 0.f;
    for (int k=0;k<Kk;++k) a += prob[h][k]*Vhs[k][t];
    att[t] = a;
  }
  __syncthreads();
  if (t < Dm){
    float o = 0.f;
    for (int j=0;j<Dm;++j) o += att[j]*Wo[j*Dm+t];
    hb[t] = curs[t] + o;
  }
  __syncthreads();
  if (t == 0){
    float s=0.f; for (int i=0;i<Dm;++i) s += hb[i];
    float m = s*(1.f/Dm);
    float v=0.f; for (int i=0;i<Dm;++i){ float d=hb[i]-m; v += d*d; }
    red[0]=m; red[1]=v*(1.f/Dm);
  }
  __syncthreads();
  if (t < Dm) hb[t] = g1[t]*(hb[t]-red[0])*rsqrtf(red[1]+1e-5f) + b1[t];
  __syncthreads();
  if (t < FFd){
    float a = bf1[t];
    for (int d=0;d<Dm;++d) a += hb[d]*W1[d*FFd+t];
    ffs[t] = fmaxf(a, 0.f);
  }
  __syncthreads();
  if (t < Dm){
    float a = bf2[t];
    for (int j=0;j<FFd;++j) a += ffs[j]*W2[j*Dm+t];
    cfp[t] = hb[t] + a;
  }
  __syncthreads();
  if (t == 0){
    float s=0.f; for (int i=0;i<Dm;++i) s += cfp[i];
    float m = s*(1.f/Dm);
    float v=0.f; for (int i=0;i<Dm;++i){ float d=cfp[i]-m; v += d*d; }
    red[0]=m; red[1]=v*(1.f/Dm);
  }
  __syncthreads();
  if (t < Dm) cfp[t] = g2[t]*(cfp[t]-red[0])*rsqrtf(red[1]+1e-5f) + b2[t];
  __syncthreads();
  if (t < Dm){
    float a=0.f;
    for (int j=0;j<Dm;++j) a += cfp[j]*Pq[j*Dm+t];
    pqs[t]=a;
  }
  for (int i=t; i<Kk*Dm; i+=512){
    int k = i >> 7, d = i & 127;
    float a=0.f;
    for (int j=0;j<Dm;++j) a += cf[k][j]*Pk[j*Dm+d];
    pks[k][d]=a;
  }
  __syncthreads();
  if (t < Kk){
    float s=0.f;
    for (int d=0;d<Dm;++d) s += pqs[d]*pks[t][d];
    s *= 0.08838834764831845f;          // 1/sqrt(128)
    s = 10.f*tanhf(s);
    if (cmv[t] == 1.f) s = -1e8f;
    uarr[t] = s;
  }
  __syncthreads();
  if (t == 0){
    float mx=-3e38f; for (int k=0;k<Kk;++k) mx = fmaxf(mx, uarr[k]);
    float sum=0.f;   for (int k=0;k<Kk;++k) sum += __expf(uarr[k]-mx);
    red[0]=mx; red[1]=logf(sum);
    float vv = bval[0];
    for (int d=0;d<Dm;++d) vv += cfp[d]*Wval[d];
    value[b] = vv;
  }
  __syncthreads();
  if (t < Kk) logp[b*Kk+t] = uarr[t] - red[0] - red[1];
}

// ---------------- launcher ----------------
extern "C" void kernel_launch(void* const* d_in, const int* in_sizes, int n_in,
                              void* d_out, int out_size, void* d_ws, size_t ws_size,
                              hipStream_t stream) {
  (void)in_sizes; (void)n_in; (void)out_size; (void)ws_size;
  // input indices (setup_inputs dict order, params flattened in insertion order)
  // 0 node,1 edge,2 budget,3 cur_idx,4 h0,5 c0,6 pos,7 mask,8 next_belief,
  // 9 We,10 be,11 Wi,12 bi,13 Wp,14 bp,15 Wb,16 bb,
  // 17..28 enc{Wq,Wk,Wv,Wo,g1,b1,W1,bf1,W2,bf2,g2,b2},
  // 29..40 dec{...}, 41 Pq,42 Pk,43 Wih,44 Whh,45 bih,46 bhh,47 Wc,48 bc,49 Wval,50 bval
  auto F = [&](int i){ return (const float*)d_in[i]; };

  char* ws = (char*)d_ws;
  const size_t SZ = (size_t)MROWS*Dm*sizeof(float);           // 52,428,800 B
  float*    bufA = (float*)(ws);                              // emb0 -> ffo -> emb2
  float*    bufB = (float*)(ws + SZ);                         // attnO -> hln
  float*    bufC = (float*)(ws + 2*SZ);                       // attn@Wo -> embenc
  _Float16* ffn1 = (_Float16*)(ws + 3*SZ);                    // MROWS*FFd f16
  _Float16* Qh   = (_Float16*)(ws + 3*SZ + (size_t)MROWS*FFd*2);
  _Float16* Ktb  = (_Float16*)((char*)Qh  + (size_t)Bsz*NH*Nn *DHd*2);
  _Float16* Vpb  = (_Float16*)((char*)Ktb + (size_t)Bsz*NH*DHd*LKP*2);
  _Float16* wf   = (_Float16*)((char*)Vpb + (size_t)Bsz*NH*LKP*DHd*2);

  // f16 weight staging: Wq,Wk,Wv,Wo @0/16384/32768/49152; W1 @65536; W2 @131072; Wb[:128] @196608
  auto cvt = [&](int idx, int off, int n){
    k_cvt<<<(n+255)/256, 256, 0, stream>>>(F(idx), wf + off, n);
  };
  cvt(17, 0,      16384); cvt(18, 16384, 16384); cvt(19, 32768, 16384); cvt(20, 49152, 16384);
  cvt(23, 65536,  65536); cvt(25, 131072, 65536);
  cvt(15, 196608, 16384);

  // zero K^T / V padded buffers (contiguous)
  size_t nz = (size_t)Bsz*NH*DHd*LKP * 2;
  k_zero_h<<<(int)((nz+255)/256), 256, 0, stream>>>(Ktb, nz);

  // 1) embedding
  k_embed<<<MROWS*Dm/256, 256, 0, stream>>>((const float*)d_in[0], F(6),
                                            F(9), F(10), F(11), F(12), F(13), F(14), bufA);
  // 2) QKV projections (head-split f16 scatter), 16x64 per wave
  k_gemm_qkv<<<1600, 256, 0, stream>>>(bufA, wf + 0,     Qh,  0);
  k_gemm_qkv<<<1600, 256, 0, stream>>>(bufA, wf + 16384, Ktb, 1);
  k_gemm_qkv<<<1600, 256, 0, stream>>>(bufA, wf + 32768, Vpb, 2);
  // 3) flash attention -> bufB
  k_attn<<<Bsz*NH*25/8, 256, 0, stream>>>(Qh, Ktb, Vpb, bufB);
  // 4) Wo -> bufC ; LN1(emb0 + attnWo) -> bufB     (dyn LDS = 2*64*32*4)
  k_gemm<float><<<1600, 256, 16384, stream>>>(bufB, wf + 49152, nullptr, bufC, nullptr, MROWS, 128, 128, 0);
  k_add_ln<<<MROWS/8, 256, 0, stream>>>(bufA, bufC, F(21), F(22), bufB);
  // 5) FFN: relu(h@W1+bf1) f16 -> ffn1 (LDS 2*16*32*4) ; @W2+bf2 -> bufA (LDS 2*64*32*2) ; LN2 -> bufC
  k_gemm<float>   <<<6400, 256, 4096, stream>>>(bufB, wf + 65536,  F(24), nullptr, ffn1, MROWS, 128, 512, 1);
  k_gemm<_Float16><<<1600, 256, 8192, stream>>>(ffn1, wf + 131072, F(26), bufA, nullptr, MROWS, 512, 128, 0);
  k_add_ln<<<MROWS/8, 256, 0, stream>>>(bufB, bufA, F(27), F(28), bufC);
  // 6) Wb projection (K=128 part via WMMA) + tail (budget & 0.4 columns) -> emb2 in bufA
  k_gemm<float><<<1600, 256, 16384, stream>>>(bufC, wf + 196608, F(16), bufA, nullptr, MROWS, 128, 128, 0);
  k_wb_tail<<<MROWS*Dm/256, 256, 0, stream>>>(bufA, F(2), F(15));

  // outputs: logp(5120), value(256), h1(32768), c1(32768)
  float* out      = (float*)d_out;
  float* out_logp = out;
  float* out_val  = out + 5120;
  float* out_h1   = out + 5376;
  float* out_c1   = out + 38144;

  // 7) LSTM
  k_lstm<<<Bsz, 512, 0, stream>>>(bufA, (const int*)d_in[3], F(8), F(4), F(5),
                                  F(43), F(44), F(45), F(46), out_h1, out_c1);
  // 8) decoder block + pointer + value
  k_decoder<<<Bsz, 512, 0, stream>>>(bufA, out_h1,
      (const int*)d_in[1], (const int*)d_in[3], F(2), (const int*)d_in[7],
      F(47), F(48),
      F(29), F(30), F(31), F(32), F(33), F(34),
      F(35), F(36), F(37), F(38), F(39), F(40),
      F(41), F(42), F(49), F(50),
      out_logp, out_val);
}
